// Attention_5265629905090
// MI455X (gfx1250) — matrix operand order repaired and checked by reading, not compiled, not run on hardware
//
#include <hip/hip_runtime.h>

typedef __attribute__((ext_vector_type(16))) __bf16        v16bf;
typedef __attribute__((ext_vector_type(8)))  float         v8f;
typedef __attribute__((ext_vector_type(4)))  unsigned int  v4u;
typedef __attribute__((ext_vector_type(8)))  int           v8i;
typedef __attribute__((ext_vector_type(4)))  int           v4i;

#define S_LEN   4096
#define DMODEL  256
#define BATCH   4
#define NEG_INF_F (-1e9f)

// Padded LDS row strides (in bf16 elements) produced by TDM pad_enable.
#define KROW 264   // 256 elems (512B) + 16B pad  -> pad_interval=6 (128 DW), pad_amount=3 (4 DW)
#define VROW 40    // 32 elems (64B)  + 16B pad  -> pad_interval=3 (16 DW),  pad_amount=3 (4 DW)

#if __has_builtin(__builtin_amdgcn_tensor_load_to_lds)
#define HAVE_TDM 1
#else
#define HAVE_TDM 0
#endif

__device__ __forceinline__ v16bf load_btile_f32(const float* __restrict__ p) {
  v16bf r;
#pragma unroll
  for (int i = 0; i < 16; ++i) r[i] = (__bf16)p[i];
  return r;
}

// ---------------------------------------------------------------------------
// Kernel 1: QKV projection.  V path emits V^T directly (A=W, B=X swap).
// ---------------------------------------------------------------------------
__global__ __launch_bounds__(128) void qkv_proj_kernel(
    const float* __restrict__ encq, const float* __restrict__ enck,
    const float* __restrict__ encv,
    const float* __restrict__ Wq, const float* __restrict__ Wk,
    const float* __restrict__ Wv,
    __bf16* __restrict__ qbuf, __bf16* __restrict__ kbuf,
    __bf16* __restrict__ vtbuf)
{
  const int wave = threadIdx.x >> 5;
  const int lane = threadIdx.x & 31;
  const int half = lane >> 4;
  const int l16  = lane & 15;
  const int gw   = blockIdx.x * 4 + wave;   // 0..12287
  const int mat  = gw >> 12;
  const int r    = gw & 4095;

  const float* Amat;
  const float* Bmat;
  int rbase, nbase;
  if (mat < 2) {
    Amat  = (mat == 0) ? encq : enck;
    Bmat  = (mat == 0) ? Wq : Wk;
    rbase = (r >> 2) * 16;
    nbase = (r & 3) * 64;
  } else {
    Amat  = Wv;
    Bmat  = encv;
    rbase = (r >> 8) * 16;
    nbase = (r & 255) * 64;
  }

  v8f acc[4];
#pragma unroll
  for (int n = 0; n < 4; ++n) acc[n] = (v8f)0.0f;

#pragma unroll
  for (int c = 0; c < 8; ++c) {
    const float* arow = Amat + (size_t)(rbase + l16) * DMODEL + 32 * c;
    v16bf at;
#pragma unroll
    for (int i = 0; i < 8; ++i) at[i]     = (__bf16)arow[16 * half + i];
#pragma unroll
    for (int i = 0; i < 8; ++i) at[8 + i] = (__bf16)arow[16 * half + 8 + i];
#pragma unroll
    for (int n = 0; n < 4; ++n) {
      const float* brow = Bmat + (size_t)(nbase + 16 * n + l16) * DMODEL + 32 * c + 16 * half;
      v16bf bt = load_btile_f32(brow);
      acc[n] = __builtin_amdgcn_wmma_f32_16x16x32_bf16(
          false, at, false, bt, (short)0, acc[n], false, false);
    }
  }

  if (mat < 2) {
    __bf16* dst = (mat == 0) ? qbuf : kbuf;
#pragma unroll
    for (int n = 0; n < 4; ++n)
#pragma unroll
      for (int v = 0; v < 8; ++v)
        dst[(size_t)(rbase + v + 8 * half) * DMODEL + nbase + 16 * n + l16] =
            (__bf16)acc[n][v];
  } else {
#pragma unroll
    for (int n = 0; n < 4; ++n) {
      const int m  = nbase + 16 * n + l16;
      const int bb = m >> 12;
      const int s  = m & (S_LEN - 1);
#pragma unroll
      for (int v = 0; v < 8; ++v) {
        const int e = rbase + v + 8 * half;
        vtbuf[((size_t)bb * DMODEL + e) * S_LEN + s] = (__bf16)acc[n][v];
      }
    }
  }
}

// ---------------------------------------------------------------------------
// TDM helper: issue a 2D tensor_load_to_lds with LDS row padding.
// D# per CDNA5 ISA §8.3/8.4 (group0: count/lds/global/type, group1: dims).
// This toolchain's builtin takes 6 args (g0, g1, g2, g3, g4, cpol).
// ---------------------------------------------------------------------------
#if HAVE_TDM
__device__ __forceinline__ void tdm_load_2d(unsigned int lds_addr,
                                            unsigned long long gaddr,
                                            int g1w0,        // data_size + pad cfg
                                            int td0, int td1,// tensor_dim0/1
                                            int tile0, int tile1,
                                            int stride0)     // tensor_dim0_stride
{
  v4u g0;
  g0[0] = 1u;                                           // count=1, user mode
  g0[1] = lds_addr;                                     // lds_addr [63:32]
  g0[2] = (unsigned int)gaddr;                          // global_addr[31:0]
  g0[3] = ((unsigned int)(gaddr >> 32) & 0x1FFFFFFu)    // global_addr[56:32]
          | (2u << 30);                                 // type = 2 ("image")
  v8i g1;
  g1[0] = g1w0;                                         // mask=0|data_size|pad
  g1[1] = td0 << 16;                                    // tensor_dim0[15:0] @ [63:48]
  g1[2] = td1 << 16;                                    // tensor_dim1[15:0] @ [95:80]
  g1[3] = tile0 << 16;                                  // tile_dim0 @ [127:112]
  g1[4] = tile1;                                        // tile_dim1 @ [143:128]
  g1[5] = stride0;                                      // dim0_stride[31:0]
  g1[6] = 0;
  g1[7] = 0;
  v4i z4 = (v4i)0;
  v8i z8 = (v8i)0;
  __builtin_amdgcn_tensor_load_to_lds(g0, g1, z4, z4, z8, 0);
}
#endif

// data_size=2B (code 1), pad_enable, pad_interval, pad_amount=3 (4 DWORDs)
#define K_G1W0 ((1 << 16) | (1 << 20) | (6 << 22) | (3 << 25))  // pad every 128 DW
#define V_G1W0 ((1 << 16) | (1 << 20) | (3 << 22) | (3 << 25))  // pad every 16 DW

// Fallback cooperative copy into the same padded LDS layout.
__device__ __forceinline__ void coop_copy_pad(__bf16* dst, const __bf16* src,
                                              int rows, int rowlen, int dstride,
                                              size_t sstride, int tid) {
  const int segs = rowlen >> 3;            // 16B segments per row
  const int total = rows * segs;
  for (int s = tid; s < total; s += 128) {
    const int rr = s / segs, cs = s % segs;
    const __bf16* sp = src + (size_t)rr * sstride + cs * 8;
    __bf16* dp = dst + rr * dstride + cs * 8;
#pragma unroll
    for (int i = 0; i < 8; ++i) dp[i] = sp[i];
  }
}

// ---------------------------------------------------------------------------
// Kernel 2: flash attention.  4 waves/block share one batch and consume
// K/V^T chunks staged in LDS by the TDM (double buffered, TENSORcnt).
// ---------------------------------------------------------------------------
__global__ __launch_bounds__(128) void flash_attn_kernel(
    const unsigned char* __restrict__ mask,
    const __bf16* __restrict__ qbuf, const __bf16* __restrict__ kbuf,
    const __bf16* __restrict__ vtbuf, float* __restrict__ out)
{
  __shared__ alignas(16) __bf16 kst[2][32 * KROW];    // 2 x 16.5 KB
  __shared__ alignas(16) __bf16 vst[2][256 * VROW];   // 2 x 20 KB
  __shared__ alignas(16) __bf16 pstage[4][16 * 40];   // per-wave P transpose

  const int wave = threadIdx.x >> 5;
  const int lane = threadIdx.x & 31;
  const int half = lane >> 4;
  const int l16  = lane & 15;
  const int b    = blockIdx.x >> 6;                   // 4 batches
  const int q0   = ((blockIdx.x & 63) * 4 + wave) * 16;

  // ---- preload Q tile (16 x 256 bf16) in A-layout registers ---------------
  v16bf qreg[8];
  {
    const __bf16* qrow = qbuf + ((size_t)b * S_LEN + q0 + l16) * DMODEL;
#pragma unroll
    for (int c = 0; c < 8; ++c) {
#pragma unroll
      for (int i = 0; i < 8; ++i) qreg[c][i]     = qrow[32 * c + 16 * half + i];
#pragma unroll
      for (int i = 0; i < 8; ++i) qreg[c][8 + i] = qrow[32 * c + 16 * half + 8 + i];
    }
  }

  v8f acc[16];
#pragma unroll
  for (int n = 0; n < 16; ++n) acc[n] = (v8f)0.0f;
  float m_run[8], l_run[8];
#pragma unroll
  for (int v = 0; v < 8; ++v) { m_run[v] = -3.0e38f; l_run[v] = 0.0f; }

  const unsigned char* mbase = mask + (size_t)b * S_LEN * S_LEN;
  __bf16* stg = &pstage[wave][0];
  const __bf16* kgbase = kbuf + (size_t)b * S_LEN * DMODEL;
  const __bf16* vgbase = vtbuf + (size_t)b * DMODEL * S_LEN;
  const int NCH = S_LEN / 32;     // 128 key chunks

#if HAVE_TDM
  if (wave == 0) {   // prefetch chunk 0 into buffer 0
    tdm_load_2d((unsigned int)(uintptr_t)&kst[0][0],
                (unsigned long long)(uintptr_t)kgbase,
                K_G1W0, 256, 16384, 256, 32, 256);
    tdm_load_2d((unsigned int)(uintptr_t)&vst[0][0],
                (unsigned long long)(uintptr_t)vgbase,
                V_G1W0, 4096, 1024, 32, 256, 4096);
  }
#endif

  for (int ch = 0; ch < NCH; ++ch) {
    const int k0  = ch * 32;
    const int cur = ch & 1;

#if HAVE_TDM
    if (wave == 0) {
      if (ch + 1 < NCH) {       // prefetch next chunk into the other buffer
        const int nxt = (ch + 1) & 1;
        tdm_load_2d((unsigned int)(uintptr_t)&kst[nxt][0],
                    (unsigned long long)(uintptr_t)(kgbase + (size_t)(k0 + 32) * DMODEL),
                    K_G1W0, 256, 16384, 256, 32, 256);
        tdm_load_2d((unsigned int)(uintptr_t)&vst[nxt][0],
                    (unsigned long long)(uintptr_t)(vgbase + (size_t)(k0 + 32)),
                    V_G1W0, 4096, 1024, 32, 256, 4096);
        __builtin_amdgcn_s_wait_tensorcnt((short)2);   // chunk ch landed
      } else {
        __builtin_amdgcn_s_wait_tensorcnt((short)0);
      }
    }
    __syncthreads();            // publish LDS chunk to all 4 waves
#else
    __syncthreads();
    coop_copy_pad(&kst[0][0], kgbase + (size_t)k0 * DMODEL, 32, 256, KROW,
                  DMODEL, (int)threadIdx.x);
    coop_copy_pad(&vst[0][0], vgbase + (size_t)k0, 256, 32, VROW,
                  S_LEN, (int)threadIdx.x);
    __syncthreads();
#endif
    const int rb = HAVE_TDM ? cur : 0;

    // ---- sims = Q @ K^T (B-tiles from LDS) --------------------------------
    v8f sims[2];
    sims[0] = (v8f)0.0f;
    sims[1] = (v8f)0.0f;
#pragma unroll
    for (int c = 0; c < 8; ++c) {
#pragma unroll
      for (int t = 0; t < 2; ++t) {
        const __bf16* kr = &kst[rb][(16 * t + l16) * KROW + 32 * c + 16 * half];
        v16bf bt;
#pragma unroll
        for (int i = 0; i < 16; ++i) bt[i] = kr[i];
        sims[t] = __builtin_amdgcn_wmma_f32_16x16x32_bf16(
            false, qreg[c], false, bt, (short)0, sims[t], false, false);
      }
    }

    // ---- scale + mask -----------------------------------------------------
#pragma unroll
    for (int t = 0; t < 2; ++t) {
      const int k = k0 + 16 * t + l16;
#pragma unroll
      for (int v = 0; v < 8; ++v) {
        const int q = q0 + v + 8 * half;
        float val = sims[t][v] * 0.0625f;
        if (mbase[(size_t)q * S_LEN + k]) val = NEG_INF_F;
        sims[t][v] = val;
      }
    }

    // ---- online softmax ---------------------------------------------------
    float nm[8];
#pragma unroll
    for (int v = 0; v < 8; ++v) nm[v] = fmaxf(sims[0][v], sims[1][v]);
#pragma unroll
    for (int off = 8; off >= 1; off >>= 1)
#pragma unroll
      for (int v = 0; v < 8; ++v) nm[v] = fmaxf(nm[v], __shfl_xor(nm[v], off, 32));

    float alpha[8], rs[8];
#pragma unroll
    for (int v = 0; v < 8; ++v) {
      const float newm = fmaxf(m_run[v], nm[v]);
      alpha[v] = __expf(m_run[v] - newm);
      m_run[v] = newm;
      const float p0 = __expf(sims[0][v] - newm);
      const float p1 = __expf(sims[1][v] - newm);
      sims[0][v] = p0;
      sims[1][v] = p1;
      rs[v] = p0 + p1;
    }
#pragma unroll
    for (int off = 8; off >= 1; off >>= 1)
#pragma unroll
      for (int v = 0; v < 8; ++v) rs[v] += __shfl_xor(rs[v], off, 32);
#pragma unroll
    for (int v = 0; v < 8; ++v) l_run[v] = l_run[v] * alpha[v] + rs[v];
#pragma unroll
    for (int n = 0; n < 16; ++n)
#pragma unroll
      for (int v = 0; v < 8; ++v) acc[n][v] *= alpha[v];

    // ---- transpose P: D-layout -> A-layout via per-wave LDS ---------------
#pragma unroll
    for (int t = 0; t < 2; ++t)
#pragma unroll
      for (int v = 0; v < 8; ++v)
        stg[(v + 8 * half) * 40 + 16 * t + l16] = (__bf16)sims[t][v];
    asm volatile("s_wait_dscnt 0" ::: "memory");
    v16bf pa;
#pragma unroll
    for (int i = 0; i < 8; ++i) pa[i]     = stg[l16 * 40 + 16 * half + i];
#pragma unroll
    for (int i = 0; i < 8; ++i) pa[8 + i] = stg[l16 * 40 + 16 * half + 8 + i];
    asm volatile("s_wait_dscnt 0" ::: "memory");

    // ---- acc += P @ V  (B-tiles = rows of V^T chunk in LDS) ---------------
#pragma unroll
    for (int n = 0; n < 16; ++n) {
      const __bf16* vr = &vst[rb][(16 * n + l16) * VROW + 16 * half];
      v16bf bt;
#pragma unroll
      for (int i = 0; i < 16; ++i) bt[i] = vr[i];
      acc[n] = __builtin_amdgcn_wmma_f32_16x16x32_bf16(
          false, pa, false, bt, (short)0, acc[n], false, false);
    }

    __syncthreads();   // all waves done with buffer rb before TDM refills it
  }

  // ---- epilogue -----------------------------------------------------------
#pragma unroll
  for (int v = 0; v < 8; ++v) {
    const float inv = 1.0f / l_run[v];
    const int q = q0 + v + 8 * half;
#pragma unroll
    for (int n = 0; n < 16; ++n)
      out[((size_t)b * S_LEN + q) * DMODEL + 16 * n + l16] = acc[n][v] * inv;
  }
}

// ---------------------------------------------------------------------------
extern "C" void kernel_launch(void* const* d_in, const int* in_sizes, int n_in,
                              void* d_out, int out_size, void* d_ws, size_t ws_size,
                              hipStream_t stream) {
  const float* encq = (const float*)d_in[0];
  const float* enck = (const float*)d_in[1];
  const float* encv = (const float*)d_in[2];
  const unsigned char* mask = (const unsigned char*)d_in[3];
  const float* Wq = (const float*)d_in[4];
  const float* Wk = (const float*)d_in[5];
  const float* Wv = (const float*)d_in[6];
  float* out = (float*)d_out;

  const size_t elems = (size_t)BATCH * S_LEN * DMODEL;
  __bf16* qbuf  = (__bf16*)d_ws;
  __bf16* kbuf  = qbuf + elems;
  __bf16* vtbuf = kbuf + elems;

  qkv_proj_kernel<<<3072, 128, 0, stream>>>(encq, enck, encv, Wq, Wk, Wv,
                                            qbuf, kbuf, vtbuf);
  flash_attn_kernel<<<256, 128, 0, stream>>>(mask, qbuf, kbuf, vtbuf, out);
}
